// AttnBlock_59700045414732
// MI455X (gfx1250) — matrix operand order repaired and checked by reading, not compiled, not run on hardware
//
#include <hip/hip_runtime.h>

// Problem constants (from reference)
#define B_    64
#define CLAT  256
#define HIDD  1024
#define DDIM  1024   // D_LAT*D_LAT
#define TDIM  1536   // TXT_H*TXT_W
#define CTXT  8

// GEMM block tiling
#define BM 128
#define BN 128
#define BK 64

typedef __attribute__((ext_vector_type(16))) __bf16 v16bf;
typedef __attribute__((ext_vector_type(8)))  float  v8f;
typedef __attribute__((ext_vector_type(4)))  int    v4i;

__device__ __forceinline__ unsigned short f32_to_bf16(float f) {
    unsigned int u = __float_as_uint(f);
    u += 0x7FFFu + ((u >> 16) & 1u);          // round-to-nearest-even
    return (unsigned short)(u >> 16);
}

union FragU { v16bf v; uint4 q[2]; };

// ---------------------------------------------------------------------------
// Async global -> LDS staging (gfx1250 GLOBAL_LOAD_ASYNC_TO_LDS_B128,
// tracked by ASYNCcnt). Fallback: synchronous bounce through registers.
#if defined(__has_builtin)
#if __has_builtin(__builtin_amdgcn_global_load_async_to_lds_b128)
#define HAVE_ASYNC_LDS 1
#endif
#endif

__device__ __forceinline__ void copy16_g2l(const unsigned short* g, unsigned short* l) {
#ifdef HAVE_ASYNC_LDS
    __builtin_amdgcn_global_load_async_to_lds_b128(
        (__attribute__((address_space(1))) v4i*)g,
        (__attribute__((address_space(3))) v4i*)l, 0, 0);
#else
    *(uint4*)l = *(const uint4*)g;
#endif
}

__device__ __forceinline__ void async_copy_wait() {
#ifdef HAVE_ASYNC_LDS
#if __has_builtin(__builtin_amdgcn_s_wait_asynccnt)
    __builtin_amdgcn_s_wait_asynccnt(0);
#else
    asm volatile("s_wait_asynccnt 0x0" ::: "memory");
#endif
#endif
}

// ---------------------------------------------------------------------------
// WMMA fragment helpers (layouts per ISA 7.12.2)

// A 16x32 bf16 from global: lane m=l&15, kh=l>>4 holds K=[k0+8kh..+7],[k0+16+8kh..+7]
__device__ __forceinline__ v16bf load_frag_a(const unsigned short* A, int lda, int row0, int k0) {
    int lane = threadIdx.x & 31;
    int m  = lane & 15;
    int kh = lane >> 4;
    const unsigned short* p = A + (size_t)(row0 + m) * lda + k0 + 8 * kh;
    FragU u;
    u.q[0] = *(const uint4*)(p);
    u.q[1] = *(const uint4*)(p + 16);
    return u.v;
}

// B 32x16 bf16 from global: lane n=l&15 holds column n, K=[k0+8kh..+7],[k0+16+8kh..+7]
// (column n of B == row n of a torch-style (out,in) weight: contiguous 32B)
__device__ __forceinline__ v16bf load_frag_b(const unsigned short* Bw, int ldb, int n0, int k0) {
    int lane = threadIdx.x & 31;
    int n  = lane & 15;
    int kh = lane >> 4;
    const unsigned short* p = Bw + (size_t)(n0 + n) * ldb + k0 + 8 * kh;
    FragU u;
    u.q[0] = *(const uint4*)(p);
    u.q[1] = *(const uint4*)(p + 16);
    return u.v;
}

// Same fragments out of an LDS tile with row stride BK
__device__ __forceinline__ v16bf frag_a_lds(const unsigned short (*T)[BK], int m0, int kk) {
    int lane = threadIdx.x & 31;
    int m  = m0 + (lane & 15);
    int kh = lane >> 4;
    FragU u;
    u.q[0] = *(const uint4*)&T[m][kk + 8 * kh];
    u.q[1] = *(const uint4*)&T[m][kk + 16 + 8 * kh];
    return u.v;
}
__device__ __forceinline__ v16bf frag_b_lds(const unsigned short (*T)[BK], int n0, int kk) {
    int lane = threadIdx.x & 31;
    int n  = n0 + (lane & 15);
    int kh = lane >> 4;
    FragU u;
    u.q[0] = *(const uint4*)&T[n][kk + 8 * kh];
    u.q[1] = *(const uint4*)&T[n][kk + 16 + 8 * kh];
    return u.v;
}

// ---------------------------------------------------------------------------
__global__ void cvt_f32_bf16_kernel(const float* __restrict__ in,
                                    unsigned short* __restrict__ out, int n) {
    int i = blockIdx.x * blockDim.x + threadIdx.x;
    if (i < n) out[i] = f32_to_bf16(in[i]);
}

// 1x1 conv over 8 text channels: t[b,o,s] = sum_c cw[o,c]*text[b,c,s] + cb[o]
__global__ void text_mix_kernel(const float* __restrict__ text,
                                const float* __restrict__ cw,
                                const float* __restrict__ cb,
                                unsigned short* __restrict__ T) {
    long long idx = (long long)blockIdx.x * blockDim.x + threadIdx.x;
    int s = (int)(idx % TDIM);
    long long r = idx / TDIM;
    int o = (int)(r % CLAT);
    int b = (int)(r / CLAT);
    const float* tb = text + (size_t)b * CTXT * TDIM;
    float acc = cb[o];
#pragma unroll
    for (int c = 0; c < CTXT; ++c)
        acc += cw[o * CTXT + c] * tb[(size_t)c * TDIM + s];
    T[idx] = f32_to_bf16(acc);
}

// ---------------------------------------------------------------------------
// LDS-tiled, double-buffered batched GEMM:
//   C[b] (256 x N) = A[b] (256 x K, bf16) * Bw^T (weight rows, bf16) + bias
// Block = 8 waves, tile BM x BN, K step BK. Wave (wm,wn) owns 64x32 of C
// (4x2 WMMA tiles). Async copies for K-tile kt+1 overlap compute on kt.
// Grid: (batch, M/BM, N/BN).
// mode 0: bf16 row-major   mode 1: bf16 transposed (N x 256)   mode 2: fp32
__global__ __launch_bounds__(256) void gemm_bf16_kernel(
    const unsigned short* __restrict__ A, unsigned long long strideA,
    const unsigned short* __restrict__ Bw,
    const float* __restrict__ bias,
    void* __restrict__ Cout, unsigned long long strideC,
    int N, int K, int mode)
{
    __shared__ unsigned short As[2][BM][BK];
    __shared__ unsigned short Bs[2][BN][BK];

    const int tid  = threadIdx.x;
    const int wave = tid >> 5;
    const int lane = tid & 31;
    const int wm   = wave >> 2;      // 0..1  (64-row halves)
    const int wn   = wave & 3;       // 0..3  (32-col strips)

    const unsigned short* Ab = A + (size_t)blockIdx.x * strideA;
    const int row0 = blockIdx.y * BM;
    const int col0 = blockIdx.z * BN;

    // stage one BM x BK A-tile and BN x BK B-tile into LDS buffer `buf`
    auto stage = [&](int buf, int k0) {
#pragma unroll
        for (int j = 0; j < 4; ++j) {
            int i = tid + 256 * j;       // uint4 index 0..1023
            int r = i >> 3;              // tile row
            int c = (i & 7) * 8;         // tile col (elements)
            copy16_g2l(Ab + (size_t)(row0 + r) * K + k0 + c, &As[buf][r][c]);
            copy16_g2l(Bw + (size_t)(col0 + r) * K + k0 + c, &Bs[buf][r][c]);
        }
    };

    const int ktiles = K / BK;
    stage(0, 0);
    async_copy_wait();
    __syncthreads();

    v8f z = {0.f,0.f,0.f,0.f,0.f,0.f,0.f,0.f};
    v8f acc[4][2] = {{z,z},{z,z},{z,z},{z,z}};

    for (int kt = 0; kt < ktiles; ++kt) {
        const int cur = kt & 1;
        if (kt + 1 < ktiles)
            stage(cur ^ 1, (kt + 1) * BK);   // async fill of next buffer

#pragma unroll
        for (int kk = 0; kk < BK; kk += 32) {
            v16bf af[4], bf[2];
#pragma unroll
            for (int mt = 0; mt < 4; ++mt)
                af[mt] = frag_a_lds(As[cur], wm * 64 + mt * 16, kk);
#pragma unroll
            for (int nt = 0; nt < 2; ++nt)
                bf[nt] = frag_b_lds(Bs[cur], wn * 32 + nt * 16, kk);
#pragma unroll
            for (int mt = 0; mt < 4; ++mt)
#pragma unroll
                for (int nt = 0; nt < 2; ++nt)
                    acc[mt][nt] = __builtin_amdgcn_wmma_f32_16x16x32_bf16(
                        false, af[mt], false, bf[nt], (short)0, acc[mt][nt], false, false);
        }
        async_copy_wait();   // our async writes to next buffer done
        __syncthreads();     // visible to all waves; prior reads retired
    }

    const int n  = lane & 15;
    const int mh = lane >> 4;
#pragma unroll
    for (int mt = 0; mt < 4; ++mt) {
#pragma unroll
        for (int nt = 0; nt < 2; ++nt) {
            int gr0 = row0 + wm * 64 + mt * 16;
            int gn  = col0 + wn * 32 + nt * 16 + n;
            float bb = bias[gn];
            if (mode == 2) {
                float* C = (float*)Cout + (size_t)blockIdx.x * strideC;
#pragma unroll
                for (int r = 0; r < 8; ++r)
                    C[(size_t)(gr0 + r + 8 * mh) * N + gn] = acc[mt][nt][r] + bb;
            } else if (mode == 1) {
                unsigned short* C = (unsigned short*)Cout + (size_t)blockIdx.x * strideC;
#pragma unroll
                for (int r = 0; r < 8; ++r)
                    C[(size_t)gn * CLAT + gr0 + r + 8 * mh] =
                        f32_to_bf16(acc[mt][nt][r] + bb);
            } else {
                unsigned short* C = (unsigned short*)Cout + (size_t)blockIdx.x * strideC;
#pragma unroll
                for (int r = 0; r < 8; ++r)
                    C[(size_t)(gr0 + r + 8 * mh) * N + gn] =
                        f32_to_bf16(acc[mt][nt][r] + bb);
            }
        }
    }
}

// ---------------------------------------------------------------------------
// Fused attention per (batch, 16-query tile), register-pipelined WMMA:
//   S = Q_tile (16x1024) * K^T  -> LDS fp32 (16x256)
//   row softmax -> bf16 probs P in LDS
//   O_tile (16x1024) = P (16x256) * V (256x1024), V given transposed (1024x256)
__global__ __launch_bounds__(256) void attn_kernel(
    const unsigned short* __restrict__ Q,
    const unsigned short* __restrict__ Kx,
    const unsigned short* __restrict__ VT,
    unsigned short* __restrict__ O)
{
    __shared__ float          S[16][256];
    __shared__ unsigned short P[16][256];
    __shared__ float          red[16][16];
    __shared__ float          rowstat[16];

    int b    = blockIdx.x;
    int qt   = blockIdx.y;
    int wave = threadIdx.x >> 5;
    int lane = threadIdx.x & 31;

    const unsigned short* Qb = Q  + (size_t)b * CLAT * HIDD;
    const unsigned short* Kb = Kx + (size_t)b * CLAT * HIDD;
    const unsigned short* Vb = VT + (size_t)b * HIDD * CLAT;

    // ---- Stage 1: scores; wave covers 32 key columns; prefetch next K-step
    {
        int n0 = wave * 32;
        v8f z = {0.f,0.f,0.f,0.f,0.f,0.f,0.f,0.f};
        v8f s0 = z, s1 = z;
        v16bf a  = load_frag_a(Qb, HIDD, qt * 16, 0);
        v16bf b0 = load_frag_b(Kb, HIDD, n0,      0);
        v16bf b1 = load_frag_b(Kb, HIDD, n0 + 16, 0);
        for (int k0 = 0; k0 < HIDD; k0 += 32) {
            int k1 = (k0 + 32) & (HIDD - 1);          // wrap: branch-free prefetch
            v16bf an  = load_frag_a(Qb, HIDD, qt * 16, k1);
            v16bf b0n = load_frag_b(Kb, HIDD, n0,      k1);
            v16bf b1n = load_frag_b(Kb, HIDD, n0 + 16, k1);
            s0 = __builtin_amdgcn_wmma_f32_16x16x32_bf16(false, a, false, b0, (short)0, s0, false, false);
            s1 = __builtin_amdgcn_wmma_f32_16x16x32_bf16(false, a, false, b1, (short)0, s1, false, false);
            a = an; b0 = b0n; b1 = b1n;
        }
        int n  = lane & 15;
        int mh = lane >> 4;
#pragma unroll
        for (int r = 0; r < 8; ++r) {
            S[r + 8 * mh][n0 + n]      = s0[r];
            S[r + 8 * mh][n0 + 16 + n] = s1[r];
        }
    }
    __syncthreads();

    // ---- Stage 2: fp32 row softmax (16 threads per row, 16 elems per thread)
    {
        int r   = threadIdx.x >> 4;
        int seg = threadIdx.x & 15;
        float mx = -1e30f;
#pragma unroll
        for (int j = 0; j < 16; ++j) mx = fmaxf(mx, S[r][seg * 16 + j]);
        red[r][seg] = mx;
        __syncthreads();
        if (threadIdx.x < 16) {
            float m = -1e30f;
#pragma unroll
            for (int j = 0; j < 16; ++j) m = fmaxf(m, red[threadIdx.x][j]);
            rowstat[threadIdx.x] = m;
        }
        __syncthreads();
        float m = rowstat[r];
        float sum = 0.f;
#pragma unroll
        for (int j = 0; j < 16; ++j) {
            float e = __expf(S[r][seg * 16 + j] - m);
            S[r][seg * 16 + j] = e;
            sum += e;
        }
        red[r][seg] = sum;
        __syncthreads();
        if (threadIdx.x < 16) {
            float s = 0.f;
#pragma unroll
            for (int j = 0; j < 16; ++j) s += red[threadIdx.x][j];
            rowstat[threadIdx.x] = 1.0f / s;
        }
        __syncthreads();
        float inv = rowstat[r];
#pragma unroll
        for (int j = 0; j < 16; ++j)
            P[r][seg * 16 + j] = f32_to_bf16(S[r][seg * 16 + j] * inv);
    }
    __syncthreads();

    // ---- Stage 3: O = P * V; wave covers 128 hidden cols; rolling B prefetch
    {
        int h0 = wave * 128;
        int m  = lane & 15;
        int kh = lane >> 4;
        v8f z = {0.f,0.f,0.f,0.f,0.f,0.f,0.f,0.f};
        v8f acc[8] = {z, z, z, z, z, z, z, z};

        auto loadA = [&](int k0) -> v16bf {
            FragU u;
            u.q[0] = *(const uint4*)&P[m][k0 + 8 * kh];
            u.q[1] = *(const uint4*)&P[m][k0 + 16 + 8 * kh];
            return u.v;
        };

        v16bf a    = loadA(0);
        v16bf bcur = load_frag_b(Vb, CLAT, h0, 0);
#pragma unroll 8
        for (int idx = 0; idx < 64; ++idx) {     // idx = (k0/32)*8 + t
            int nidx = (idx + 1) & 63;
            v16bf bnext = load_frag_b(Vb, CLAT, h0 + 16 * (nidx & 7), (nidx >> 3) * 32);
            int t = idx & 7;
            acc[t] = __builtin_amdgcn_wmma_f32_16x16x32_bf16(
                false, a, false, bcur, (short)0, acc[t], false, false);
            if (t == 7) a = loadA((nidx >> 3) * 32);
            bcur = bnext;
        }

        unsigned short* Ob = O + (size_t)b * CLAT * HIDD;
        int n  = lane & 15;
        int mh = lane >> 4;
#pragma unroll
        for (int t = 0; t < 8; ++t)
#pragma unroll
            for (int r = 0; r < 8; ++r)
                Ob[(size_t)(qt * 16 + r + 8 * mh) * HIDD + h0 + 16 * t + n] =
                    f32_to_bf16(acc[t][r]);
    }
}

// ---------------------------------------------------------------------------
extern "C" void kernel_launch(void* const* d_in, const int* in_sizes, int n_in,
                              void* d_out, int out_size, void* d_ws, size_t ws_size,
                              hipStream_t stream) {
    const float* latent = (const float*)d_in[0];
    const float* text   = (const float*)d_in[1];
    const float* conv_w = (const float*)d_in[2];
    const float* conv_b = (const float*)d_in[3];
    const float* q_w    = (const float*)d_in[4];
    const float* q_b    = (const float*)d_in[5];
    const float* k_w    = (const float*)d_in[6];
    const float* k_b    = (const float*)d_in[7];
    const float* v_w    = (const float*)d_in[8];
    const float* v_b    = (const float*)d_in[9];
    const float* out_w  = (const float*)d_in[10];
    const float* out_b  = (const float*)d_in[11];

    // Workspace carve (bf16 = unsigned short), ~228 MB total
    unsigned short* p = (unsigned short*)d_ws;
    unsigned short* qw_bf = p;  p += (size_t)HIDD * DDIM;
    unsigned short* kw_bf = p;  p += (size_t)HIDD * TDIM;
    unsigned short* vw_bf = p;  p += (size_t)HIDD * TDIM;
    unsigned short* ow_bf = p;  p += (size_t)DDIM * HIDD;
    unsigned short* l_bf  = p;  p += (size_t)B_ * CLAT * DDIM;
    unsigned short* t_bf  = p;  p += (size_t)B_ * CLAT * TDIM;
    unsigned short* q_bf  = p;  p += (size_t)B_ * CLAT * HIDD;
    unsigned short* k_bf  = p;  p += (size_t)B_ * CLAT * HIDD;
    unsigned short* vT_bf = p;  p += (size_t)B_ * HIDD * CLAT;
    unsigned short* o_bf  = p;  p += (size_t)B_ * CLAT * HIDD;

    auto cvt = [&](const float* src, unsigned short* dst, int n) {
        cvt_f32_bf16_kernel<<<(n + 255) / 256, 256, 0, stream>>>(src, dst, n);
    };
    cvt(q_w,   qw_bf, HIDD * DDIM);
    cvt(k_w,   kw_bf, HIDD * TDIM);
    cvt(v_w,   vw_bf, HIDD * TDIM);
    cvt(out_w, ow_bf, DDIM * HIDD);
    cvt(latent, l_bf, B_ * CLAT * DDIM);

    text_mix_kernel<<<(B_ * CLAT * TDIM) / 256, 256, 0, stream>>>(
        text, conv_w, conv_b, t_bf);

    dim3 gg(B_, CLAT / BM, HIDD / BN);   // (64, 2, 8)

    // Q = l * q_w^T + q_b
    gemm_bf16_kernel<<<gg, 256, 0, stream>>>(
        l_bf, (unsigned long long)CLAT * DDIM, qw_bf, q_b,
        q_bf, (unsigned long long)CLAT * HIDD, HIDD, DDIM, 0);
    // K = t * k_w^T + k_b
    gemm_bf16_kernel<<<gg, 256, 0, stream>>>(
        t_bf, (unsigned long long)CLAT * TDIM, kw_bf, k_b,
        k_bf, (unsigned long long)CLAT * HIDD, HIDD, TDIM, 0);
    // V^T = (t * v_w^T + v_b)^T   (transposed store for contiguous PV B-rows)
    gemm_bf16_kernel<<<gg, 256, 0, stream>>>(
        t_bf, (unsigned long long)CLAT * TDIM, vw_bf, v_b,
        vT_bf, (unsigned long long)HIDD * CLAT, HIDD, TDIM, 1);

    // softmax(Q K^T) V
    attn_kernel<<<dim3(B_, CLAT / 16), 256, 0, stream>>>(q_bf, k_bf, vT_bf, o_bf);

    // out = o * out_w^T + out_b  (fp32 store straight into d_out)
    gemm_bf16_kernel<<<gg, 256, 0, stream>>>(
        o_bf, (unsigned long long)CLAT * HIDD, ow_bf, out_b,
        d_out, (unsigned long long)CLAT * DDIM, DDIM, HIDD, 2);
}
